// SCNN_Layer_23673859736408
// MI455X (gfx1250) — matrix operand order repaired and checked by reading, not compiled, not run on hardware
//
#include <hip/hip_runtime.h>
#include <hip/hip_bf16.h>

// ---------------------------------------------------------------------------
// Spatial-RNN conv layer for MI455X (gfx1250, wave32, WMMA).
// out[b,co,h,w] = relu( bias[co] + sum_{ci,k} K[ci,k,co] *
//                       (x[b,ci,h,w+k-4] + out[b,ci,h-1,w+k-4]) )
// Serial over h (carry), GEMM per row via v_wmma_f32_16x16x32_f16.
// ---------------------------------------------------------------------------

typedef _Float16 v16h __attribute__((ext_vector_type(16)));
typedef _Float16 v8h  __attribute__((ext_vector_type(8)));
typedef float    v8f  __attribute__((ext_vector_type(8)));

#define B_   64
#define C_   128
#define H_   36
#define W_   100
#define KSZ  9
#define PADW 4
#define WT   32            // output w-tile per block (2 WMMA n-subtiles)
#define JR   (WT + 2*PADW) // 40 input rows in the LDS window
#define STR  136           // LDS row stride in halves (pad to kill bank conflicts)

// ---- weight repack: kernel[ci][k][0][co] (f32) -> Wh[k][co][ci] (f16) ------
__global__ __launch_bounds__(256)
void scnn_prep_weights(const float* __restrict__ kern, _Float16* __restrict__ wh)
{
    int idx = blockIdx.x * blockDim.x + threadIdx.x;   // over KSZ*C_*C_
    if (idx >= KSZ * C_ * C_) return;
    int ci = idx % C_;
    int t  = idx / C_;     // t = k*C_ + co
    int co = t % C_;
    int k  = t / C_;
    wh[idx] = (_Float16)kern[((size_t)ci * KSZ + k) * C_ + co];
}

// ---- one scan step: conv row h with carry from row h-1 ---------------------
__global__ __launch_bounds__(256)
void scnn_row(const float* __restrict__ x, const _Float16* __restrict__ wh,
              const float* __restrict__ bias, float* __restrict__ out, int h)
{
    __shared__ _Float16 s_in[JR * STR];   // [j][ci], j maps w = w0-4+j

    const int b   = blockIdx.y;
    const int w0  = blockIdx.x * WT;
    const int tid = threadIdx.x;

    // ---- stage input window + carry into LDS (f32 -> f16) ----
    const float* xrow = x   + (((size_t)b * C_) * H_ + h) * W_;       // + ci*H_*W_
    const float* prow = (h > 0)
        ? out + (((size_t)b * C_) * H_ + (h - 1)) * W_ : nullptr;

    for (int idx = tid; idx < C_ * JR; idx += 256) {
        int ci = idx / JR;
        int j  = idx % JR;
        int w  = w0 - PADW + j;
        float v = 0.0f;
        if (w >= 0 && w < W_) {
            size_t off = (size_t)ci * (H_ * W_) + w;
            v = xrow[off];
            if (prow) v += prow[off];     // carry = previous row's ReLU output
        }
        s_in[j * STR + ci] = (_Float16)v;
    }
    __syncthreads();

    const int lane = tid & 31;
    const int wv   = tid >> 5;        // wave id -> co tile
    const int lh   = (lane >> 4) & 1; // lane half (K-half selector)
    const int ln   = lane & 15;       // M row (A) / N col (B,D)
    const int co0  = wv * 16;

    v8f acc0 = {};                    // n = w0 + 0..15
    v8f acc1 = {};                    // n = w0 + 16..31

    // out[co, w] = sum_k sum_ci Wh[k][co][ci] * s_in[(w - w0 + 4) + k - 4][ci]
    for (int k = 0; k < KSZ; ++k) {
        const _Float16* wk = wh + ((size_t)k * C_ + (co0 + ln)) * C_;
        for (int c = 0; c < 4; ++c) {          // K chunks of 32 over ci
            // A fragment (16x32 f16): lane = M; 16 contiguous K at lh*16
            union { v16h v; v8h hh[2]; } a;
            a.hh[0] = *(const v8h*)(wk + c * 32 + lh * 16);
            a.hh[1] = *(const v8h*)(wk + c * 32 + lh * 16 + 8);

            // B fragment (32x16 f16): lane-half = K half, 16 contiguous K/lane
            {
                const _Float16* bp = &s_in[(ln + k) * STR + c * 32 + lh * 16];
                union { v16h v; v8h hh[2]; } bb;
                bb.hh[0] = *(const v8h*)(bp);
                bb.hh[1] = *(const v8h*)(bp + 8);
                acc0 = __builtin_amdgcn_wmma_f32_16x16x32_f16(
                    false, a.v, false, bb.v, (short)0, acc0, false, false);
            }
            {
                const _Float16* bp = &s_in[(16 + ln + k) * STR + c * 32 + lh * 16];
                union { v16h v; v8h hh[2]; } bb;
                bb.hh[0] = *(const v8h*)(bp);
                bb.hh[1] = *(const v8h*)(bp + 8);
                acc1 = __builtin_amdgcn_wmma_f32_16x16x32_f16(
                    false, a.v, false, bb.v, (short)0, acc1, false, false);
            }
        }
    }

    // ---- epilogue: bias + ReLU + guarded stores ----
    // D layout: VGPR r -> M = r + 8*lh, N = ln
    for (int r = 0; r < 8; ++r) {
        int co = co0 + r + 8 * lh;
        float bv = bias[co];
        size_t base = (((size_t)b * C_ + co) * H_ + h) * W_;
        int wa = w0 + ln;
        int wb = w0 + 16 + ln;
        float va = acc0[r] + bv; va = va > 0.0f ? va : 0.0f;
        float vb = acc1[r] + bv; vb = vb > 0.0f ? vb : 0.0f;
        if (wa < W_) out[base + wa] = va;
        if (wb < W_) out[base + wb] = vb;
    }
}

extern "C" void kernel_launch(void* const* d_in, const int* in_sizes, int n_in,
                              void* d_out, int out_size, void* d_ws, size_t ws_size,
                              hipStream_t stream)
{
    const float* x    = (const float*)d_in[0];   // (64,128,36,100) f32
    const float* kern = (const float*)d_in[1];   // (128,9,1,128)   f32
    const float* bias = (const float*)d_in[2];   // (128,)          f32
    float*       out  = (float*)d_out;           // (64,128,36,100) f32
    _Float16*    wh   = (_Float16*)d_ws;         // 9*128*128 f16 = 288 KB

    // repack weights once per call (L2-resident, reused by all 36 row launches)
    int nw = KSZ * C_ * C_;
    scnn_prep_weights<<<(nw + 255) / 256, 256, 0, stream>>>(kern, wh);

    // serial scan over H: each row reads the previous row's output as carry
    dim3 grid((W_ + WT - 1) / WT, B_);   // (4, 64)
    for (int h = 0; h < H_; ++h)
        scnn_row<<<grid, 256, 0, stream>>>(x, wh, bias, out, h);
}
